// EarthAttention3D_76888504533460
// MI455X (gfx1250) — compile-verified, matrix operand order checked by reading, not run on hardware
//
#include <hip/hip_runtime.h>
#include <hip/hip_bf16.h>

// ---------------------------------------------------------------------------
// EarthAttention3D fused kernel for gfx1250 (MI455X).
//  Shapes: x[30,64,144,192] f32, qkv_w[576,192], proj_w[192,192],
//          bias_table[3312,64,6], pos_index[144,144] i32. Out f32 same as x.
//  1 workgroup (8 wave32) per (b,w) window; bf16 WMMA 16x16x32 / f32 accum;
//  all intermediates LDS-resident (268 KB of the 320 KB/WG budget);
//  weights pre-packed into per-lane WMMA fragment order in d_ws.
//  All tile indices are readfirstlane'd: they are wave-uniform by
//  construction but LLVM cannot prove it, and without this the epilogues
//  compile to divergent EXEC-masked stores (seen in round 1).
// ---------------------------------------------------------------------------

typedef __attribute__((ext_vector_type(16))) __bf16 v16bf;
typedef __attribute__((ext_vector_type(8)))  float  v8f;

#define NB     30
#define NWIN   64
#define NTOK   144
#define CDIM   192
#define C3     576
#define NHEAD  6
#define HD     32
#define NPAD   160                   // 144 keys padded to 5 K-chunks of 32
#define SCALE  0.17677669529663687f  // 32^-0.5

#define QKV_FRAGS  (36 * 6)          // 36 col-tiles x 6 K-chunks
#define PROJ_FRAGS (12 * 6)
#define PROJ_FRAG_OFF (QKV_FRAGS * 512)   // elements (frag = 32 lanes x 16 bf16)

static __device__ inline v8f vzero8() {
    v8f z = {0.f, 0.f, 0.f, 0.f, 0.f, 0.f, 0.f, 0.f};
    return z;
}

// ---------------------------------------------------------------------------
// Prolog: pack f32 weights into bf16 WMMA-B fragment order.
//   lane l holds B[k][n] for n = tile*16 + (l&15), k = kc*32 + (l>>4)*16 + i,
//   i = 0..15, stored contiguously -> 32B per lane per fragment.
// B[k][d] = w[d][k]  =>  each lane reads 16 contiguous f32 from one w row.
// ---------------------------------------------------------------------------
__global__ __launch_bounds__(256) void pack_weights(
        const float* __restrict__ qkv_w,    // [576,192] row-major
        const float* __restrict__ proj_w,   // [192,192] row-major
        __bf16* __restrict__ ws) {
    int t = blockIdx.x * 256 + threadIdx.x;
    if (t >= (QKV_FRAGS + PROJ_FRAGS) * 32) return;
    int frag = t >> 5, lane = t & 31;
    const float* src;
    __bf16* dst;
    int ct, kc;
    if (frag < QKV_FRAGS) {
        ct = frag / 6; kc = frag % 6;
        src = qkv_w;  dst = ws + (size_t)frag * 512;
    } else {
        int f2 = frag - QKV_FRAGS;
        ct = f2 / 6;  kc = f2 % 6;
        src = proj_w; dst = ws + PROJ_FRAG_OFF + (size_t)f2 * 512;
    }
    int d  = ct * 16 + (lane & 15);
    int kb = kc * 32 + ((lane >> 4) << 4);
    const float* row = src + (size_t)d * CDIM + kb;
    #pragma unroll
    for (int i = 0; i < 16; ++i)
        dst[lane * 16 + i] = (__bf16)row[i];
}

// ---------------------------------------------------------------------------
// Main fused kernel.
// ---------------------------------------------------------------------------
__global__ __launch_bounds__(256) void earth_attn(
        const float* __restrict__ x,          // [1920,144,192]
        const float* __restrict__ qkv_b,      // [576]
        const float* __restrict__ proj_b,     // [192]
        const float* __restrict__ bias_table, // [3312,64,6]
        const int*   __restrict__ pos_index,  // [144,144]
        const __bf16* __restrict__ wfrag,     // packed weights in d_ws
        float* __restrict__ out) {

    __shared__ __align__(64) __bf16 s_x [NTOK * CDIM];   // x (ph1 A); attn-out (ph2/3)
    __shared__ __align__(64) __bf16 s_q [NTOK * CDIM];   // scaled q, row-major
    __shared__ __align__(64) __bf16 s_k [NTOK * CDIM];   // k, row-major (B of q@k^T)
    __shared__ __align__(64) __bf16 s_vt[CDIM * NPAD];   // v transposed [chan][key]
    __shared__ __align__(64) __bf16 s_p [8 * 16 * NPAD]; // per-wave P strips [16][160]

    const int tid   = threadIdx.x;
    const int lane  = tid & 31;
    const int waveU = __builtin_amdgcn_readfirstlane(tid >> 5);  // wave-uniform
    const int bw    = blockIdx.x;        // (b*64 + w)
    const int w     = bw & (NWIN - 1);
    const int lr    = lane & 15;         // lane row/col within tile
    const int lh    = lane >> 4;         // K-half / row-half selector
    const int rowPhase = lh << 3;        // +8 rows for upper half-wave

    // ---- phase 0: stage x as bf16; zero K-padding regions -----------------
    {
        const float4* xg = (const float4*)(x + (size_t)bw * NTOK * CDIM);
        for (int i = tid; i < NTOK * CDIM / 4; i += 256) {
            float4 v = xg[i];
            s_x[4 * i + 0] = (__bf16)v.x;
            s_x[4 * i + 1] = (__bf16)v.y;
            s_x[4 * i + 2] = (__bf16)v.z;
            s_x[4 * i + 3] = (__bf16)v.w;
        }
        for (int i = tid; i < CDIM * 16; i += 256)       // v^T keys 144..159 = 0
            s_vt[(i >> 4) * NPAD + NTOK + (i & 15)] = (__bf16)0.f;
        for (int i = tid; i < 8 * 16 * 16; i += 256)     // P cols 144..159 = 0
            s_p[(i >> 4) * NPAD + NTOK + (i & 15)] = (__bf16)0.f;
    }
    __syncthreads();

    // ---- phase 1: QKV = X @ Wqkv^T + b  --------------------------------
    // Three separate tile loops (q / k / v) so each epilogue is branch-free.
    // 108 tiles each (9 row-tiles x 12 col-tiles), K=192 in 6 chunks.

    // --- q tiles: global col tiles 0..11 -> s_q (scaled) ---
    for (int t = waveU; t < 9 * 12; t += 8) {
        int tU = __builtin_amdgcn_readfirstlane(t);
        int rt = tU / 12, ct = tU % 12;
        v8f acc = vzero8();
        #pragma unroll
        for (int kc = 0; kc < 6; ++kc) {
            v16bf a = *(const v16bf*)&s_x[(rt * 16 + lr) * CDIM + kc * 32 + lh * 16];
            v16bf b = *(const v16bf*)&wfrag[((size_t)(ct * 6 + kc) * 32 + lane) * 16];
            acc = __builtin_amdgcn_wmma_f32_16x16x32_bf16(false, a, false, b,
                                                          (short)0, acc, false, false);
        }
        int   rem = ct * 16 + lr;
        float bq  = qkv_b[rem];
        #pragma unroll
        for (int i = 0; i < 8; ++i)
            s_q[(rt * 16 + i + rowPhase) * CDIM + rem] = (__bf16)((acc[i] + bq) * SCALE);
    }
    // --- k tiles: global col tiles 12..23 -> s_k ---
    for (int t = waveU; t < 9 * 12; t += 8) {
        int tU = __builtin_amdgcn_readfirstlane(t);
        int rt = tU / 12, ct = tU % 12;
        v8f acc = vzero8();
        #pragma unroll
        for (int kc = 0; kc < 6; ++kc) {
            v16bf a = *(const v16bf*)&s_x[(rt * 16 + lr) * CDIM + kc * 32 + lh * 16];
            v16bf b = *(const v16bf*)&wfrag[((size_t)((12 + ct) * 6 + kc) * 32 + lane) * 16];
            acc = __builtin_amdgcn_wmma_f32_16x16x32_bf16(false, a, false, b,
                                                          (short)0, acc, false, false);
        }
        int   rem = ct * 16 + lr;
        float bq  = qkv_b[192 + rem];
        #pragma unroll
        for (int i = 0; i < 8; ++i)
            s_k[(rt * 16 + i + rowPhase) * CDIM + rem] = (__bf16)(acc[i] + bq);
    }
    // --- v tiles: global col tiles 24..35 -> s_vt (transposed) ---
    for (int t = waveU; t < 9 * 12; t += 8) {
        int tU = __builtin_amdgcn_readfirstlane(t);
        int rt = tU / 12, ct = tU % 12;
        v8f acc = vzero8();
        #pragma unroll
        for (int kc = 0; kc < 6; ++kc) {
            v16bf a = *(const v16bf*)&s_x[(rt * 16 + lr) * CDIM + kc * 32 + lh * 16];
            v16bf b = *(const v16bf*)&wfrag[((size_t)((24 + ct) * 6 + kc) * 32 + lane) * 16];
            acc = __builtin_amdgcn_wmma_f32_16x16x32_bf16(false, a, false, b,
                                                          (short)0, acc, false, false);
        }
        int   rem = ct * 16 + lr;      // v channel
        float bq  = qkv_b[384 + rem];
        #pragma unroll
        for (int i = 0; i < 8; ++i)
            s_vt[rem * NPAD + (rt * 16 + i + rowPhase)] = (__bf16)(acc[i] + bq);
    }
    __syncthreads();

    // ---- phase 2: attention per (head, 16-row strip) ----------------------
    for (int s = waveU; s < NHEAD * 9; s += 8) {
        int sU = __builtin_amdgcn_readfirstlane(s);
        int h = sU / 9, rt = sU % 9;
        v8f accs[9];
        v16bf aq = *(const v16bf*)&s_q[(rt * 16 + lr) * CDIM + h * HD + lh * 16];
        #pragma unroll
        for (int t = 0; t < 9; ++t) {     // scores: K = HD = 32, one WMMA each
            v16bf bk = *(const v16bf*)&s_k[(t * 16 + lr) * CDIM + h * HD + lh * 16];
            accs[t] = __builtin_amdgcn_wmma_f32_16x16x32_bf16(false, aq, false, bk,
                                                              (short)0, vzero8(), false, false);
        }
        // earth-position bias gather (table + index both L2-resident)
        #pragma unroll
        for (int t = 0; t < 9; ++t) {
            int col = t * 16 + lr;
            #pragma unroll
            for (int i = 0; i < 8; ++i) {
                int row = rt * 16 + i + rowPhase;
                int idx = pos_index[row * NTOK + col];
                accs[t][i] += bias_table[((size_t)idx * NWIN + w) * NHEAD + h];
            }
        }
        // row softmax: each row's 144 cols live across one 16-lane half
        float rmax[8], rinv[8];
        #pragma unroll
        for (int i = 0; i < 8; ++i) {
            float m = -3.0e38f;
            #pragma unroll
            for (int t = 0; t < 9; ++t) m = fmaxf(m, accs[t][i]);
            #pragma unroll
            for (int o = 1; o < 16; o <<= 1) m = fmaxf(m, __shfl_xor(m, o, 16));
            rmax[i] = m;
        }
        #pragma unroll
        for (int i = 0; i < 8; ++i) {
            float sum = 0.f;
            #pragma unroll
            for (int t = 0; t < 9; ++t) {
                float e = __expf(accs[t][i] - rmax[i]);
                accs[t][i] = e;
                sum += e;
            }
            #pragma unroll
            for (int o = 1; o < 16; o <<= 1) sum += __shfl_xor(sum, o, 16);
            rinv[i] = 1.f / sum;
        }
        // D-layout -> A-layout via per-wave LDS strip (cols 144..159 stay 0)
        const int pbase = waveU * 16 * NPAD;
        #pragma unroll
        for (int t = 0; t < 9; ++t) {
            #pragma unroll
            for (int i = 0; i < 8; ++i)
                s_p[pbase + (i + rowPhase) * NPAD + t * 16 + lr] =
                    (__bf16)(accs[t][i] * rinv[i]);
        }
        // out strip = P[16,160] @ V[160, HD]  (5 K-chunks, 2 col tiles)
        #pragma unroll
        for (int ct = 0; ct < 2; ++ct) {
            v8f acc = vzero8();
            #pragma unroll
            for (int kc = 0; kc < 5; ++kc) {
                v16bf a = *(const v16bf*)&s_p[pbase + lr * NPAD + kc * 32 + lh * 16];
                v16bf b = *(const v16bf*)&s_vt[(h * HD + ct * 16 + lr) * NPAD + kc * 32 + lh * 16];
                acc = __builtin_amdgcn_wmma_f32_16x16x32_bf16(false, a, false, b,
                                                              (short)0, acc, false, false);
            }
            #pragma unroll
            for (int i = 0; i < 8; ++i)
                s_x[(rt * 16 + i + rowPhase) * CDIM + h * HD + ct * 16 + lr] =
                    (__bf16)acc[i];                        // reuse s_x as obuf
        }
    }
    __syncthreads();

    // ---- phase 3: projection: out = attnOut @ Wproj^T + b -----------------
    const __bf16* pw = wfrag + PROJ_FRAG_OFF;
    float* ob = out + (size_t)bw * NTOK * CDIM;
    for (int t = waveU; t < 9 * 12; t += 8) {
        int tU = __builtin_amdgcn_readfirstlane(t);
        int rt = tU / 12, ct = tU % 12;
        v8f acc = vzero8();
        #pragma unroll
        for (int kc = 0; kc < 6; ++kc) {
            v16bf a = *(const v16bf*)&s_x[(rt * 16 + lr) * CDIM + kc * 32 + lh * 16];
            v16bf b = *(const v16bf*)&pw[((size_t)(ct * 6 + kc) * 32 + lane) * 16];
            acc = __builtin_amdgcn_wmma_f32_16x16x32_bf16(false, a, false, b,
                                                          (short)0, acc, false, false);
        }
        int   col = ct * 16 + lr;
        float pb  = proj_b[col];
        #pragma unroll
        for (int i = 0; i < 8; ++i)
            ob[(size_t)(rt * 16 + i + rowPhase) * CDIM + col] = acc[i] + pb;
    }
}

// ---------------------------------------------------------------------------
extern "C" void kernel_launch(void* const* d_in, const int* in_sizes, int n_in,
                              void* d_out, int out_size, void* d_ws, size_t ws_size,
                              hipStream_t stream) {
    const float* x      = (const float*)d_in[0];
    const float* qkv_w  = (const float*)d_in[1];
    const float* qkv_b  = (const float*)d_in[2];
    const float* proj_w = (const float*)d_in[3];
    const float* proj_b = (const float*)d_in[4];
    const float* btab   = (const float*)d_in[5];
    const int*   pidx   = (const int*)d_in[6];
    __bf16* ws = (__bf16*)d_ws;   // needs (216+72)*512*2 = ~295 KB scratch

    int packThreads = (QKV_FRAGS + PROJ_FRAGS) * 32;   // 9216
    pack_weights<<<(packThreads + 255) / 256, 256, 0, stream>>>(qkv_w, proj_w, ws);
    earth_attn<<<NB * NWIN, 256, 0, stream>>>(x, qkv_b, proj_b, btab, pidx, ws,
                                              (float*)d_out);
}
